// MemBottleNeck_46119358825052
// MI455X (gfx1250) — compile-verified, matrix operand order checked by reading, not run on hardware
//
#include <hip/hip_runtime.h>
#include <hip/hip_bf16.h>
#include <math.h>

// ---------------------------------------------------------------------------
// MemAE forward pass for MI455X (gfx1250, wave32, WMMA).
//   B=1024, C=64, F=16 -> D=16384, MID=2048, LAT=16, MEM=25
// Dominant GEMMs (x@W1^T, h2@W4^T: 137 GFLOP, ~400MB HBM) run on
// v_wmma_f32_16x16x32_bf16 with fp32->bf16 conversion done ONCE per weight
// tile in LDS (shared by all 4 waves of the block), double-buffered and
// software-pipelined so global loads overlap WMMA issue.
// ---------------------------------------------------------------------------

#define B_    1024
#define D_    16384
#define MID_  2048
#define LAT_  16
#define NMEM_ 25
#define RCHUNK_ 16            // BN stats row chunks

typedef __attribute__((ext_vector_type(16))) __bf16 v16bf;
typedef __attribute__((ext_vector_type(8)))  float  v8f;

// Raw (pre-conversion) fragment data: two contiguous 8-float K-runs per lane.
struct Raw { float4 a, b, c, d; };

__device__ __forceinline__ Raw load_raw(const float* __restrict__ p) {
  const float4* q = (const float4*)p;
  Raw r;
  r.a = q[0];   // K + 0..3
  r.b = q[1];   // K + 4..7
  r.c = q[4];   // K + 16..19
  r.d = q[5];   // K + 20..23
  return r;
}

__device__ __forceinline__ v16bf cvt_raw(const Raw& r) {
  v16bf f;
  f[0]  = (__bf16)r.a.x; f[1]  = (__bf16)r.a.y; f[2]  = (__bf16)r.a.z; f[3]  = (__bf16)r.a.w;
  f[4]  = (__bf16)r.b.x; f[5]  = (__bf16)r.b.y; f[6]  = (__bf16)r.b.z; f[7]  = (__bf16)r.b.w;
  f[8]  = (__bf16)r.c.x; f[9]  = (__bf16)r.c.y; f[10] = (__bf16)r.c.z; f[11] = (__bf16)r.c.w;
  f[12] = (__bf16)r.d.x; f[13] = (__bf16)r.d.y; f[14] = (__bf16)r.d.z; f[15] = (__bf16)r.d.w;
  return f;
}

// ---------------------------------------------------------------------------
// Y[M,N] = X[M,K] @ W[N,K]^T + bias[N]   (fp32 in memory, bf16 MACs, f32 acc)
// Block = 128 threads = 4 waves; block tile 64(M) x 64(N).
// Wave w computes rows m0+16w..+15 across all 64 N columns (4 WMMA / K-step).
// B (weight) fragments are converted once per block into LDS (fragment
// layout: 32 contiguous bytes per lane) and double-buffered; A and W raw
// loads are prefetched one K-step ahead. M%64==0, N%64==0, K%64==0 assumed.
// ---------------------------------------------------------------------------
__global__ __launch_bounds__(128)
void gemm_xwt_bias_wmma(const float* __restrict__ X, const float* __restrict__ W,
                        const float* __restrict__ bias, float* __restrict__ Y,
                        int M, int N, int K) {
  const int t    = threadIdx.x;
  const int lane = t & 31;
  const int wave = t >> 5;
  const int row  = lane & 15;
  const int half = lane >> 4;
  const int m0   = blockIdx.y * 64 + wave * 16;
  const int n0   = blockIdx.x * 64;

  // 2 buffers x 4 subtiles x 32 lanes x 16 bf16 (32B per lane slot) = 8 KB
  __shared__ alignas(32) __bf16 ldsB[2][4 * 32 * 16];

  // W-staging role: this thread produces subtile (t>>5), lane (t&31)'s slot.
  const float* wp = W + (size_t)(n0 + (t >> 5) * 16 + (t & 15)) * K
                      + (((t & 31) >> 4) << 3);
  __bf16* slot0 = &ldsB[0][((t >> 5) * 32 + (t & 31)) * 16];
  __bf16* slot1 = &ldsB[1][((t >> 5) * 32 + (t & 31)) * 16];

  const float* xp = X + (size_t)(m0 + row) * K + (half << 3);

  v8f acc0 = {}; v8f acc1 = {}; v8f acc2 = {}; v8f acc3 = {};

  // Prologue: fill buffer 0 with k=0 W tile, prefetch k=0 A raw.
  Raw wr = load_raw(wp);
  Raw ar = load_raw(xp);
  *(v16bf*)slot0 = cvt_raw(wr);
  __syncthreads();

  int p = 0;
  for (int k0 = 0; k0 < K; k0 += 32) {
    const bool more = (k0 + 32) < K;
    Raw wn, an;
    if (more) {                      // issue next K-step's global loads early
      wn = load_raw(wp + k0 + 32);
      an = load_raw(xp + k0 + 32);
    }

    const v16bf a = cvt_raw(ar);
    const __bf16* base = &ldsB[p][lane * 16];
    const v16bf b0 = *(const v16bf*)(base + 0 * 512);
    const v16bf b1 = *(const v16bf*)(base + 1 * 512);
    const v16bf b2 = *(const v16bf*)(base + 2 * 512);
    const v16bf b3 = *(const v16bf*)(base + 3 * 512);

    acc0 = __builtin_amdgcn_wmma_f32_16x16x32_bf16(false, a, false, b0, (short)0, acc0, false, false);
    acc1 = __builtin_amdgcn_wmma_f32_16x16x32_bf16(false, a, false, b1, (short)0, acc1, false, false);
    acc2 = __builtin_amdgcn_wmma_f32_16x16x32_bf16(false, a, false, b2, (short)0, acc2, false, false);
    acc3 = __builtin_amdgcn_wmma_f32_16x16x32_bf16(false, a, false, b3, (short)0, acc3, false, false);

    if (more) {
      *(v16bf*)(p ? slot0 : slot1) = cvt_raw(wn);   // fill other buffer
      ar = an;
    }
    __syncthreads();   // all reads of buf p done; buf p^1 filled
    p ^= 1;
  }

  // C/D layout (ISA 7.12.2): elem r -> row = m0 + r + 8*half, col = n0+16t+(lane&15)
  const int mr = m0 + (half << 3);
  const int nc = lane & 15;
  const float bv0 = bias[n0 +  0 + nc];
  const float bv1 = bias[n0 + 16 + nc];
  const float bv2 = bias[n0 + 32 + nc];
  const float bv3 = bias[n0 + 48 + nc];
#pragma unroll
  for (int r = 0; r < 8; ++r) {
    float* yr = Y + (size_t)(mr + r) * N + n0 + nc;
    yr[0]  = acc0[r] + bv0;
    yr[16] = acc1[r] + bv1;
    yr[32] = acc2[r] + bv2;
    yr[48] = acc3[r] + bv3;
  }
}

// ---------------------------------------------------------------------------
// BatchNorm column stats, two-stage: 16 row-chunk partials then finalize.
// ---------------------------------------------------------------------------
__global__ __launch_bounds__(256)
void bn_stats_partial(const float* __restrict__ Yv, float* __restrict__ psum,
                      float* __restrict__ psq) {
  const int c  = blockIdx.x * 256 + threadIdx.x;
  const int ch = blockIdx.y;
  const int r0 = ch * (B_ / RCHUNK_);
  float s = 0.f, s2 = 0.f;
  for (int r = r0; r < r0 + B_ / RCHUNK_; ++r) {
    const float v = Yv[(size_t)r * MID_ + c];
    s += v; s2 += v * v;
  }
  psum[ch * MID_ + c] = s;
  psq [ch * MID_ + c] = s2;
}

__global__ __launch_bounds__(256)
void bn_stats_final(const float* __restrict__ psum, const float* __restrict__ psq,
                    float* __restrict__ mean, float* __restrict__ rstd) {
  const int c = blockIdx.x * 256 + threadIdx.x;
  float s = 0.f, s2 = 0.f;
#pragma unroll
  for (int i = 0; i < RCHUNK_; ++i) { s += psum[i * MID_ + c]; s2 += psq[i * MID_ + c]; }
  const float mu  = s  * (1.f / B_);
  const float var = s2 * (1.f / B_) - mu * mu;   // biased variance
  mean[c] = mu;
  rstd[c] = rsqrtf(var + 1e-5f);
}

__global__ __launch_bounds__(256)
void bn_apply_relu(const float* __restrict__ Yv, const float* __restrict__ mean,
                   const float* __restrict__ rstd, const float* __restrict__ g,
                   const float* __restrict__ be, float* __restrict__ H) {
  const size_t i = (size_t)blockIdx.x * 256 + threadIdx.x;
  const int c = (int)(i & (MID_ - 1));
  const float v = (Yv[i] - mean[c]) * rstd[c] * g[c] + be[c];
  H[i] = fmaxf(v, 0.f);
}

// ---------------------------------------------------------------------------
// wave32 butterfly reductions
// ---------------------------------------------------------------------------
__device__ __forceinline__ float wave_sum(float v) {
#pragma unroll
  for (int o = 16; o > 0; o >>= 1) v += __shfl_xor(v, o, 32);
  return v;
}
__device__ __forceinline__ float wave_max(float v) {
#pragma unroll
  for (int o = 16; o > 0; o >>= 1) v = fmaxf(v, __shfl_xor(v, o, 32));
  return v;
}

// ---------------------------------------------------------------------------
// Memory module: one wave per batch row.
// ---------------------------------------------------------------------------
__global__ __launch_bounds__(256)
void mem_module(const float* __restrict__ h, const float* __restrict__ W2,
                const float* __restrict__ b2, const float* __restrict__ memW,
                float* __restrict__ z, float* __restrict__ att,
                float* __restrict__ zhat) {
  const int lane = threadIdx.x & 31;
  const int wave = threadIdx.x >> 5;
  const int r    = blockIdx.x * 8 + wave;

  __shared__ float zsh[8][LAT_];
  __shared__ float ash[8][32];

  float acc[LAT_];
#pragma unroll
  for (int j = 0; j < LAT_; ++j) acc[j] = 0.f;

  const float* hr = h + (size_t)r * MID_;
  for (int k = lane; k < MID_; k += 32) {
    const float hv = hr[k];
#pragma unroll
    for (int j = 0; j < LAT_; ++j) acc[j] += hv * W2[j * MID_ + k];
  }
#pragma unroll
  for (int j = 0; j < LAT_; ++j) acc[j] = wave_sum(acc[j]);

  if (lane == 0) {
#pragma unroll
    for (int j = 0; j < LAT_; ++j) {
      const float zv = acc[j] + b2[j];
      z[(size_t)r * LAT_ + j] = zv;
      zsh[wave][j] = zv;
    }
  }
  __syncthreads();

  float tv = -1e30f;
  if (lane < NMEM_) {
    tv = 0.f;
#pragma unroll
    for (int j = 0; j < LAT_; ++j) tv += zsh[wave][j] * memW[lane * LAT_ + j];
  }
  const float mx = wave_max(tv);
  const float e  = (lane < NMEM_) ? __expf(tv - mx) : 0.f;
  const float se = wave_sum(e);
  const float a0 = e / se;
  const float s  = a0 - 0.0025f;
  float av = fmaxf(s, 0.f) * a0 / (fabsf(s) + 1e-12f);
  const float l1 = wave_sum((lane < NMEM_) ? fabsf(av) : 0.f);
  av = av / fmaxf(l1, 1e-12f);
  if (lane < NMEM_) {
    att[(size_t)r * NMEM_ + lane] = av;
    ash[wave][lane] = av;
  }
  __syncthreads();

  if (lane < LAT_) {
    float zh = 0.f;
#pragma unroll
    for (int m = 0; m < NMEM_; ++m) zh += ash[wave][m] * memW[m * LAT_ + lane];
    zhat[(size_t)r * LAT_ + lane] = zh;
  }
}

// ---------------------------------------------------------------------------
// Decoder fc1: y3[B,MID] = z_hat[B,LAT] @ W3[MID,LAT]^T + b3   (K=16, tiny)
// ---------------------------------------------------------------------------
__global__ __launch_bounds__(256)
void dec_fc1(const float* __restrict__ zhat, const float* __restrict__ W3,
             const float* __restrict__ b3, float* __restrict__ Yv) {
  const size_t i = (size_t)blockIdx.x * 256 + threadIdx.x;
  const int r = (int)(i >> 11);            // / MID_
  const int n = (int)(i & (MID_ - 1));     // % MID_
  const float* zr = zhat + (size_t)r * LAT_;
  const float* wr = W3 + (size_t)n * LAT_;
  float s = b3[n];
#pragma unroll
  for (int k = 0; k < LAT_; ++k) s += zr[k] * wr[k];
  Yv[i] = s;
}

// ---------------------------------------------------------------------------
extern "C" void kernel_launch(void* const* d_in, const int* in_sizes, int n_in,
                              void* d_out, int out_size, void* d_ws, size_t ws_size,
                              hipStream_t stream) {
  (void)in_sizes; (void)n_in; (void)out_size; (void)ws_size;

  const float* x    = (const float*)d_in[0];
  const float* W1   = (const float*)d_in[1];
  const float* b1   = (const float*)d_in[2];
  const float* g1   = (const float*)d_in[3];
  const float* be1  = (const float*)d_in[4];
  const float* W2   = (const float*)d_in[5];
  const float* b2   = (const float*)d_in[6];
  const float* memW = (const float*)d_in[7];
  const float* W3   = (const float*)d_in[8];
  const float* b3   = (const float*)d_in[9];
  const float* g2   = (const float*)d_in[10];
  const float* be2  = (const float*)d_in[11];
  const float* W4   = (const float*)d_in[12];
  const float* b4   = (const float*)d_in[13];

  // Output tuple (out, att, z, z_hat) concatenated flat.
  float* out  = (float*)d_out;
  float* att  = out + (size_t)B_ * D_;
  float* z    = att + (size_t)B_ * NMEM_;
  float* zhat = z   + (size_t)B_ * LAT_;

  // Workspace layout (~17.1 MB).
  float* bufA = (float*)d_ws;                 // [B, MID] pre-BN
  float* bufB = bufA + (size_t)B_ * MID_;     // [B, MID] post-BN/ReLU
  float* mean = bufB + (size_t)B_ * MID_;
  float* rstd = mean + MID_;
  float* psum = rstd + MID_;                  // [RCHUNK_, MID]
  float* psq  = psum + (size_t)RCHUNK_ * MID_;

  const dim3 blk128(128), blk256(256);
  const dim3 gStat(MID_ / 256, RCHUNK_);

  // Encoder fc1 (dominant GEMM #1): y1 = x @ W1^T + b1
  gemm_xwt_bias_wmma<<<dim3(MID_ / 64, B_ / 64), blk128, 0, stream>>>(
      x, W1, b1, bufA, B_, MID_, D_);
  bn_stats_partial<<<gStat, blk256, 0, stream>>>(bufA, psum, psq);
  bn_stats_final<<<dim3(MID_ / 256), blk256, 0, stream>>>(psum, psq, mean, rstd);
  bn_apply_relu<<<dim3((B_ * MID_) / 256), blk256, 0, stream>>>(
      bufA, mean, rstd, g1, be1, bufB);

  // Bottleneck + memory addressing (writes z, att, z_hat to output)
  mem_module<<<dim3(B_ / 8), blk256, 0, stream>>>(bufB, W2, b2, memW, z, att, zhat);

  // Decoder fc1 + BN + ReLU
  dec_fc1<<<dim3((B_ * MID_) / 256), blk256, 0, stream>>>(zhat, W3, b3, bufA);
  bn_stats_partial<<<gStat, blk256, 0, stream>>>(bufA, psum, psq);
  bn_stats_final<<<dim3(MID_ / 256), blk256, 0, stream>>>(psum, psq, mean, rstd);
  bn_apply_relu<<<dim3((B_ * MID_) / 256), blk256, 0, stream>>>(
      bufA, mean, rstd, g2, be2, bufB);

  // Decoder fc2 (dominant GEMM #2): out = h2 @ W4^T + b4
  gemm_xwt_bias_wmma<<<dim3(D_ / 64, B_ / 64), blk128, 0, stream>>>(
      bufB, W4, b4, out, B_, D_, MID_);
}
